// GaussianBlurConv_83949430767737
// MI455X (gfx1250) — compile-verified, matrix operand order checked by reading, not run on hardware
//
#include <hip/hip_runtime.h>
#include <hip/hip_bf16.h>
#include <cstdint>

typedef __bf16 bf16_t;
typedef __attribute__((ext_vector_type(16))) __bf16 v16bf;
typedef __attribute__((ext_vector_type(8)))  float  v8f;

#define LDT 72   // LDS row stride in bf16 elems: 64 data + 8 pad -> 144B rows (16B aligned)

#if defined(__AMDGCN__) && __has_builtin(__builtin_amdgcn_global_load_async_to_lds_b128) && __has_builtin(__builtin_amdgcn_s_wait_asynccnt)
#define USE_ASYNC_LDS 1
#else
#define USE_ASYNC_LDS 0
#endif

// The async-to-LDS builtin takes pointers to GCC-vector int4 in AS1 (global)
// and AS3 (LDS), per the clang diagnostic from the probe round.
typedef int vint4 __attribute__((vector_size(16)));
typedef __attribute__((address_space(1))) vint4 gvint4;
typedef __attribute__((address_space(3))) vint4 lvint4;

union FragU { v16bf v; unsigned u[8]; };
union AccU  { v8f  v; float f[8]; };
union Pk8   { uint4 q; bf16_t h[8]; };

// 16-byte global -> LDS copy. On gfx1250 this is a VGPR-free async DMA
// (GLOBAL_LOAD_ASYNC_TO_LDS_B128, tracked by ASYNCcnt).
__device__ __forceinline__ void async_cp16(const bf16_t* g, bf16_t* l) {
#if USE_ASYNC_LDS
  __builtin_amdgcn_global_load_async_to_lds_b128((gvint4*)(g), (lvint4*)(l), 0, 0);
#else
  *(uint4*)l = *(const uint4*)g;
#endif
}

__device__ __forceinline__ void wait_async() {
#if USE_ASYNC_LDS
  __builtin_amdgcn_s_wait_asynccnt(0);
#endif
}

// Load one 16x32 bf16 WMMA operand fragment from an LDS tile (row-major, K
// contiguous, row stride LDT). Matches CDNA5 16-bit A/B layout:
//   lanes 0-15: rows 0-15, K 0-7 in v0-3 (2/dword), K 16-23 in v4-7
//   lanes 16-31: rows 0-15, K 8-15 / 24-31
__device__ __forceinline__ v16bf ld_frag(const bf16_t* p0, int lane) {
  const int r   = lane & 15;
  const int kg2 = (lane >> 4) << 2;           // dword offset of K-group: 0 or 4
  const unsigned* p = (const unsigned*)(p0 + r * LDT);
  FragU f;
#pragma unroll
  for (int i = 0; i < 4; ++i) {
    f.u[i]     = p[kg2 + i];
    f.u[4 + i] = p[8 + kg2 + i];
  }
  return f.v;
}

// ---------------- fp32 -> bf16 elementwise convert ----------------
__global__ __launch_bounds__(256) void cvt_bf16(const float* __restrict__ in,
                                                bf16_t* __restrict__ out, int n) {
  size_t i = ((size_t)blockIdx.x * 256 + threadIdx.x) * 8;
  if (i + 8 <= (size_t)n) {
    float4 a = *(const float4*)(in + i);
    float4 b = *(const float4*)(in + i + 4);
    Pk8 o;
    o.h[0] = (bf16_t)a.x; o.h[1] = (bf16_t)a.y; o.h[2] = (bf16_t)a.z; o.h[3] = (bf16_t)a.w;
    o.h[4] = (bf16_t)b.x; o.h[5] = (bf16_t)b.y; o.h[6] = (bf16_t)b.z; o.h[7] = (bf16_t)b.w;
    *(uint4*)(out + i) = o.q;
  }
}

// ---------------- fp32 [R][C] -> bf16 [C][R] transpose convert ----------------
__global__ __launch_bounds__(256) void transpose_cvt(const float* __restrict__ in,
                                                     bf16_t* __restrict__ out,
                                                     int R, int C) {
  __shared__ float tile[32][33];
  const int c0 = blockIdx.x * 32, r0 = blockIdx.y * 32;
  const int tx = threadIdx.x, ty = threadIdx.y;   // block (32,8)
#pragma unroll
  for (int i = 0; i < 32; i += 8)
    tile[ty + i][tx] = in[(size_t)(r0 + ty + i) * C + (c0 + tx)];
  __syncthreads();
#pragma unroll
  for (int i = 0; i < 32; i += 8)
    out[(size_t)(c0 + ty + i) * R + (r0 + tx)] = (bf16_t)tile[tx][ty + i];
}

// ---------------- GEMM1: relu(x @ W1 + b1) -> bf16 ----------------
// A [M][Kd] bf16 row-major, Bt [N][Kd] bf16 (W1 transposed, N-major)
__global__ __launch_bounds__(256) void gemm1_bias_relu(
    const bf16_t* __restrict__ A, const bf16_t* __restrict__ Bt,
    const float* __restrict__ bias, bf16_t* __restrict__ C, int N, int Kd) {
  __shared__ __align__(16) bf16_t sA[2][64 * LDT];
  __shared__ __align__(16) bf16_t sB[2][128 * LDT];

  const int tid  = threadIdx.x;
  const int lane = tid & 31;
  const int wave = tid >> 5;
  const int wm = wave >> 2;                 // 0..1  (rows)
  const int wn = wave & 3;                  // 0..3  (cols)
  const size_t bm = (size_t)blockIdx.y * 64;
  const size_t bn = (size_t)blockIdx.x * 128;

  AccU acc[2][2];
#pragma unroll
  for (int i = 0; i < 2; ++i)
#pragma unroll
    for (int j = 0; j < 2; ++j)
#pragma unroll
      for (int r = 0; r < 8; ++r) acc[i][j].f[r] = 0.f;

  const int nstage = Kd >> 6;

  // per-thread staging slices: A 2 x 16B, B 4 x 16B per stage
  const int vA0 = tid,            rA0 = vA0 >> 3, kA0 = (vA0 & 7) * 8;
  const int vA1 = tid + 256,      rA1 = vA1 >> 3, kA1 = (vA1 & 7) * 8;
  const bf16_t* gA0 = A + (bm + rA0) * (size_t)Kd + kA0;
  const bf16_t* gA1 = A + (bm + rA1) * (size_t)Kd + kA1;

  const bf16_t* gB[4];
  int rB[4], kB[4];
#pragma unroll
  for (int i = 0; i < 4; ++i) {
    int v = tid + i * 256; rB[i] = v >> 3; kB[i] = (v & 7) * 8;
    gB[i] = Bt + (bn + rB[i]) * (size_t)Kd + kB[i];
  }

  // stage 0 -> buffer 0
  async_cp16(gA0, &sA[0][rA0 * LDT + kA0]);
  async_cp16(gA1, &sA[0][rA1 * LDT + kA1]);
#pragma unroll
  for (int i = 0; i < 4; ++i) async_cp16(gB[i], &sB[0][rB[i] * LDT + kB[i]]);
  wait_async();
  __syncthreads();

  for (int s = 0; s < nstage; ++s) {
    const int cur = s & 1;
    if (s + 1 < nstage) {                    // async DMA of next stage -> other buffer
      const size_t k0 = (size_t)(s + 1) << 6;
      const int nxt = cur ^ 1;
      async_cp16(gA0 + k0, &sA[nxt][rA0 * LDT + kA0]);
      async_cp16(gA1 + k0, &sA[nxt][rA1 * LDT + kA1]);
#pragma unroll
      for (int i = 0; i < 4; ++i) async_cp16(gB[i] + k0, &sB[nxt][rB[i] * LDT + kB[i]]);
    }
#pragma unroll
    for (int ks = 0; ks < 64; ks += 32) {
      v16bf a0 = ld_frag(&sA[cur][(wm * 32 +  0) * LDT + ks], lane);
      v16bf a1 = ld_frag(&sA[cur][(wm * 32 + 16) * LDT + ks], lane);
      v16bf b0 = ld_frag(&sB[cur][(wn * 32 +  0) * LDT + ks], lane);
      v16bf b1 = ld_frag(&sB[cur][(wn * 32 + 16) * LDT + ks], lane);
      acc[0][0].v = __builtin_amdgcn_wmma_f32_16x16x32_bf16(false, a0, false, b0, (short)0, acc[0][0].v, false, false);
      acc[0][1].v = __builtin_amdgcn_wmma_f32_16x16x32_bf16(false, a0, false, b1, (short)0, acc[0][1].v, false, false);
      acc[1][0].v = __builtin_amdgcn_wmma_f32_16x16x32_bf16(false, a1, false, b0, (short)0, acc[1][0].v, false, false);
      acc[1][1].v = __builtin_amdgcn_wmma_f32_16x16x32_bf16(false, a1, false, b1, (short)0, acc[1][1].v, false, false);
    }
    if (s + 1 < nstage) {
      wait_async();
      __syncthreads();
    }
  }

  // epilogue: bias + relu, bf16 out. C/D layout: vgpr r -> M = r + 8*(lane>=16), N = lane&15
  const int nl = lane & 15;
  const int mo = (lane >> 4) << 3;
#pragma unroll
  for (int sm = 0; sm < 2; ++sm)
#pragma unroll
    for (int sn = 0; sn < 2; ++sn) {
      const size_t n = bn + wn * 32 + sn * 16 + nl;
      const float bv = bias[n];
#pragma unroll
      for (int r = 0; r < 8; ++r) {
        const size_t m = bm + wm * 32 + sm * 16 + mo + r;
        float v = acc[sm][sn].f[r] + bv;
        C[m * (size_t)N + n] = (bf16_t)(v > 0.f ? v : 0.f);
      }
    }
}

// ---------------- in-place row LayerNorm on bf16 [rows][2048] ----------------
__global__ __launch_bounds__(256) void ln_rows_bf16(bf16_t* __restrict__ H,
                                                    const float* __restrict__ g,
                                                    const float* __restrict__ be, int N) {
  bf16_t* p = H + (size_t)blockIdx.x * N;
  const int tid = threadIdx.x, lane = tid & 31, wave = tid >> 5;
  Pk8 in;
  in.q = *(const uint4*)(p + tid * 8);
  float v[8], s = 0.f, s2 = 0.f;
#pragma unroll
  for (int i = 0; i < 8; ++i) { v[i] = (float)in.h[i]; s += v[i]; s2 += v[i] * v[i]; }
#pragma unroll
  for (int off = 16; off; off >>= 1) { s += __shfl_xor(s, off, 32); s2 += __shfl_xor(s2, off, 32); }
  __shared__ float red[16];
  if (lane == 0) { red[wave] = s; red[8 + wave] = s2; }
  __syncthreads();
  float ts = 0.f, ts2 = 0.f;
#pragma unroll
  for (int w = 0; w < 8; ++w) { ts += red[w]; ts2 += red[8 + w]; }
  const float mean = ts / (float)N;
  const float var  = ts2 / (float)N - mean * mean;
  const float rstd = rsqrtf(var + 1e-5f);
  Pk8 out;
#pragma unroll
  for (int i = 0; i < 8; ++i) {
    const int c = tid * 8 + i;
    out.h[i] = (bf16_t)((v[i] - mean) * rstd * g[c] + be[c]);
  }
  *(uint4*)(p + tid * 8) = out.q;
}

// -------- GEMM2 + bias + relu + LN(64) + (.W3 + b3) + sigmoid -> sigma --------
// A [B][Kd] bf16, Bt [64][Kd] bf16 (W2 transposed)
__global__ __launch_bounds__(256) void gemm2_head(
    const bf16_t* __restrict__ A, const bf16_t* __restrict__ Bt,
    const float* __restrict__ b2, const float* __restrict__ g2,
    const float* __restrict__ be2, const float* __restrict__ W3,
    const float* __restrict__ b3, float* __restrict__ sigma, int Kd) {
  __shared__ __align__(16) bf16_t sA[2][128 * LDT];
  __shared__ __align__(16) bf16_t sB[2][64 * LDT];
  __shared__ float hs[128 * 68];

  const int tid  = threadIdx.x;
  const int lane = tid & 31;
  const int wave = tid >> 5;
  const int wm = wave >> 1;                 // 0..3
  const int wn = wave & 1;                  // 0..1
  const size_t bm = (size_t)blockIdx.x * 128;

  AccU acc[2][2];
#pragma unroll
  for (int i = 0; i < 2; ++i)
#pragma unroll
    for (int j = 0; j < 2; ++j)
#pragma unroll
      for (int r = 0; r < 8; ++r) acc[i][j].f[r] = 0.f;

  const int nstage = Kd >> 6;

  const bf16_t* gA[4];
  int rA[4], kA[4];
#pragma unroll
  for (int i = 0; i < 4; ++i) {
    int v = tid + i * 256; rA[i] = v >> 3; kA[i] = (v & 7) * 8;
    gA[i] = A + (bm + rA[i]) * (size_t)Kd + kA[i];
  }
  const int vB0 = tid,       rB0 = vB0 >> 3, kB0 = (vB0 & 7) * 8;
  const int vB1 = tid + 256, rB1 = vB1 >> 3, kB1 = (vB1 & 7) * 8;
  const bf16_t* gB0 = Bt + (size_t)rB0 * Kd + kB0;
  const bf16_t* gB1 = Bt + (size_t)rB1 * Kd + kB1;

#pragma unroll
  for (int i = 0; i < 4; ++i) async_cp16(gA[i], &sA[0][rA[i] * LDT + kA[i]]);
  async_cp16(gB0, &sB[0][rB0 * LDT + kB0]);
  async_cp16(gB1, &sB[0][rB1 * LDT + kB1]);
  wait_async();
  __syncthreads();

  for (int s = 0; s < nstage; ++s) {
    const int cur = s & 1;
    if (s + 1 < nstage) {
      const size_t k0 = (size_t)(s + 1) << 6;
      const int nxt = cur ^ 1;
#pragma unroll
      for (int i = 0; i < 4; ++i) async_cp16(gA[i] + k0, &sA[nxt][rA[i] * LDT + kA[i]]);
      async_cp16(gB0 + k0, &sB[nxt][rB0 * LDT + kB0]);
      async_cp16(gB1 + k0, &sB[nxt][rB1 * LDT + kB1]);
    }
#pragma unroll
    for (int ks = 0; ks < 64; ks += 32) {
      v16bf a0 = ld_frag(&sA[cur][(wm * 32 +  0) * LDT + ks], lane);
      v16bf a1 = ld_frag(&sA[cur][(wm * 32 + 16) * LDT + ks], lane);
      v16bf b0 = ld_frag(&sB[cur][(wn * 32 +  0) * LDT + ks], lane);
      v16bf b1 = ld_frag(&sB[cur][(wn * 32 + 16) * LDT + ks], lane);
      acc[0][0].v = __builtin_amdgcn_wmma_f32_16x16x32_bf16(false, a0, false, b0, (short)0, acc[0][0].v, false, false);
      acc[0][1].v = __builtin_amdgcn_wmma_f32_16x16x32_bf16(false, a0, false, b1, (short)0, acc[0][1].v, false, false);
      acc[1][0].v = __builtin_amdgcn_wmma_f32_16x16x32_bf16(false, a1, false, b0, (short)0, acc[1][0].v, false, false);
      acc[1][1].v = __builtin_amdgcn_wmma_f32_16x16x32_bf16(false, a1, false, b1, (short)0, acc[1][1].v, false, false);
    }
    if (s + 1 < nstage) {
      wait_async();
      __syncthreads();
    }
  }

  // park relu(h+b2) in LDS, one 128x64 tile per block
  const int nl = lane & 15;
  const int mo = (lane >> 4) << 3;
#pragma unroll
  for (int sm = 0; sm < 2; ++sm)
#pragma unroll
    for (int sn = 0; sn < 2; ++sn) {
      const int n = wn * 32 + sn * 16 + nl;
      const float bv = b2[n];
#pragma unroll
      for (int r = 0; r < 8; ++r) {
        const int m = wm * 32 + sm * 16 + mo + r;
        float v = acc[sm][sn].f[r] + bv;
        hs[m * 68 + n] = v > 0.f ? v : 0.f;
      }
    }
  __syncthreads();

  // two threads per row: LN over 64, dot with W3, sigmoid
  const int row = tid >> 1, half = tid & 1;
  const float* hr = hs + row * 68 + half * 32;
  float s = 0.f, s2 = 0.f;
#pragma unroll
  for (int j = 0; j < 32; ++j) { float v = hr[j]; s += v; s2 += v * v; }
  s  += __shfl_xor(s, 1, 32);
  s2 += __shfl_xor(s2, 1, 32);
  const float mean = s * (1.f / 64.f);
  const float var  = s2 * (1.f / 64.f) - mean * mean;
  const float rstd = rsqrtf(var + 1e-5f);
  float dot = 0.f;
#pragma unroll
  for (int j = 0; j < 32; ++j) {
    const int c = half * 32 + j;
    dot += ((hr[j] - mean) * rstd * g2[c] + be2[c]) * W3[c];
  }
  dot += __shfl_xor(dot, 1, 32);
  if (half == 0) {
    const float z = dot + b3[0];
    sigma[bm + row] = 1.f / (1.f + __expf(-z)) + 1e-10f;
  }
}

// ---------------- per-row 5-tap gaussian blur ----------------
__global__ __launch_bounds__(256) void gauss_conv(const float* __restrict__ x,
                                                  const float* __restrict__ sigma,
                                                  float* __restrict__ out, int S) {
  __shared__ float tile[1024 + 4];
  const int row = blockIdx.y;
  const int s0  = blockIdx.x * 1024;
  const float* xr = x + (size_t)row * S;
  const int tid = threadIdx.x;
  for (int i = tid; i < 1024 + 4; i += 256) {
    const int gx = s0 - 2 + i;
    tile[i] = (gx >= 0 && gx < S) ? xr[gx] : 0.f;
  }
  const float sg  = sigma[row];
  const float inv = 1.f / (sg * sg);
  const float e1 = __expf(-0.5f * inv);
  const float e2 = __expf(-2.0f * inv);
  const float nm = 1.f / (1.f + 2.f * e1 + 2.f * e2);
  const float w0 = e2 * nm, w1 = e1 * nm, w2 = nm;
  __syncthreads();
  const int b = tid * 4;
  const float a0 = tile[b], a1 = tile[b + 1], a2 = tile[b + 2], a3 = tile[b + 3],
              a4 = tile[b + 4], a5 = tile[b + 5], a6 = tile[b + 6], a7 = tile[b + 7];
  float4 o;
  o.x = w0 * a0 + w1 * a1 + w2 * a2 + w1 * a3 + w0 * a4;
  o.y = w0 * a1 + w1 * a2 + w2 * a3 + w1 * a4 + w0 * a5;
  o.z = w0 * a2 + w1 * a3 + w2 * a4 + w1 * a5 + w0 * a6;
  o.w = w0 * a3 + w1 * a4 + w2 * a5 + w1 * a6 + w0 * a7;
  *(float4*)(out + (size_t)row * S + s0 + b) = o;
}

extern "C" void kernel_launch(void* const* d_in, const int* in_sizes, int n_in,
                              void* d_out, int out_size, void* d_ws, size_t ws_size,
                              hipStream_t stream) {
  (void)in_sizes; (void)n_in; (void)out_size; (void)ws_size;
  const int Bq = 8192, S = 4096, H1 = 2048, H2 = 64;

  const float* x   = (const float*)d_in[0];
  const float* W1  = (const float*)d_in[1];
  const float* b1  = (const float*)d_in[2];
  const float* g1  = (const float*)d_in[3];
  const float* be1 = (const float*)d_in[4];
  const float* W2  = (const float*)d_in[5];
  const float* b2  = (const float*)d_in[6];
  const float* g2  = (const float*)d_in[7];
  const float* be2 = (const float*)d_in[8];
  const float* W3  = (const float*)d_in[9];
  const float* b3  = (const float*)d_in[10];
  float* out = (float*)d_out;
  char* ws = (char*)d_ws;

  size_t off = 0;
  auto take = [&](size_t bytes) { size_t r = off; off += (bytes + 255) & ~(size_t)255; return r; };
  bf16_t* xb   = (bf16_t*)(ws + take((size_t)Bq * S * 2));
  bf16_t* W1t  = (bf16_t*)(ws + take((size_t)S * H1 * 2));
  bf16_t* W2t  = (bf16_t*)(ws + take((size_t)H1 * H2 * 2));
  bf16_t* h1b  = (bf16_t*)(ws + take((size_t)Bq * H1 * 2));
  float*  sig  = (float*)(ws + take((size_t)Bq * 4));

  const int nX = Bq * S;
  cvt_bf16<<<nX / (8 * 256), 256, 0, stream>>>(x, xb, nX);

  dim3 tb(32, 8);
  transpose_cvt<<<dim3(H1 / 32, S / 32), tb, 0, stream>>>(W1, W1t, S, H1);
  transpose_cvt<<<dim3(H2 / 32, H1 / 32), tb, 0, stream>>>(W2, W2t, H1, H2);

  gemm1_bias_relu<<<dim3(H1 / 128, Bq / 64), 256, 0, stream>>>(xb, W1t, b1, h1b, H1, S);
  ln_rows_bf16<<<Bq, 256, 0, stream>>>(h1b, g1, be1, H1);
  gemm2_head<<<Bq / 128, 256, 0, stream>>>(h1b, W2t, b2, g2, be2, W3, b3, sig, H1);
  gauss_conv<<<dim3(S / 1024, Bq), 256, 0, stream>>>(x, sig, out, S);
}
